// MultiMemoryInfiniAttention_9698036154870
// MI455X (gfx1250) — matrix operand order repaired and checked by reading, not compiled, not run on hardware
//
#include <hip/hip_runtime.h>
#include <hip/hip_bf16.h>

typedef __attribute__((ext_vector_type(16))) __bf16          v16bf;
typedef __attribute__((ext_vector_type(8)))  float           v8f;
typedef __attribute__((ext_vector_type(16))) unsigned short  v16u;
typedef __attribute__((ext_vector_type(8)))  unsigned short  v8u;

constexpr int S  = 2048;
constexpr int H  = 1024;
constexpr int NH = 16;
constexpr int D  = 64;
constexpr int NM = 4;
constexpr float EPS = 1e-6f;

__device__ __forceinline__ unsigned short f2bf(float f) {
  unsigned u = __float_as_uint(f);
  u += 0x7FFFu + ((u >> 16) & 1u);          // round-to-nearest-even
  return (unsigned short)(u >> 16);
}
__device__ __forceinline__ float bf2f(unsigned short h) {
  return __uint_as_float(((unsigned)h) << 16);
}

union Frag16 { v16u u16; v8u h[2]; v16bf bf; };

// A fragment 16x32 bf16 from row-major [row][k] (k contiguous), ld in elements.
__device__ __forceinline__ v16bf load_a(const unsigned short* base, size_t row_off,
                                        int ld, int k0, int lane) {
  const int half = lane >> 4, m = lane & 15;
  const unsigned short* p = base + row_off + (size_t)m * ld + k0 + half * 8;
  Frag16 f;
  f.h[0] = *(const v8u*)p;          // K = k0 + 8*half .. +7
  f.h[1] = *(const v8u*)(p + 16);   // K = k0 + 16 + 8*half .. +7
  return f.bf;
}

// B fragment 32x16 bf16 from B-col layout [n][k] (k contiguous), ld in elements.
__device__ __forceinline__ v16bf load_b(const unsigned short* base, size_t n_row_off,
                                        int ld, int k0, int lane) {
  const int half = lane >> 4, n = lane & 15;
  const unsigned short* p = base + n_row_off + (size_t)n * ld + k0 + half * 8;
  Frag16 f;
  f.h[0] = *(const v8u*)p;          // K = k0 + 8*half .. +7
  f.h[1] = *(const v8u*)(p + 16);   // K = k0 + 16 + 8*half .. +7
  return f.bf;
}

__device__ __forceinline__ v8f wmma_bf16(v16bf a, v16bf b, v8f c) {
  return __builtin_amdgcn_wmma_f32_16x16x32_bf16(false, a, false, b, (short)0, c,
                                                 false, false);
}

// ---------------------------------------------------------------- converts
__global__ void k_f32_to_bf16(const float* __restrict__ in,
                              unsigned short* __restrict__ out, int n) {
  int i = blockIdx.x * blockDim.x + threadIdx.x;
  if (i < n) out[i] = f2bf(in[i]);
}

// memories [m,h,d,e] -> bf16 memT [m,h,e,d]  (B-col layout for WMMA)
__global__ void k_memT(const float* __restrict__ mem, unsigned short* __restrict__ out) {
  int i = blockIdx.x * blockDim.x + threadIdx.x;
  if (i < NM * NH * D * D) {
    int e = i & 63, d = (i >> 6) & 63, mh = i >> 12;
    out[((size_t)mh * D + e) * D + d] = f2bf(mem[i]);
  }
}

// ---------------------------------------------------------------- QKV GEMM
// y = X @ W.T ; z=0: phi->sq, z=1: phi->sk & skT, z=2: v -> vT
__global__ __launch_bounds__(256) void k_qkv(
    const unsigned short* __restrict__ Xb,
    const unsigned short* __restrict__ Wqb, const unsigned short* __restrict__ Wkb,
    const unsigned short* __restrict__ Wvb,
    unsigned short* __restrict__ sqb, unsigned short* __restrict__ skb,
    unsigned short* __restrict__ skTb, unsigned short* __restrict__ vTb) {
  const int lane = threadIdx.x & 31, wave = threadIdx.x >> 5;
  const int s0 = blockIdx.x * 16;
  const int o0 = blockIdx.y * 128 + wave * 16;
  const int z  = blockIdx.z;
  const unsigned short* Wb = (z == 0) ? Wqb : (z == 1) ? Wkb : Wvb;
  v8f acc = {};
  for (int k0 = 0; k0 < H; k0 += 32) {
    v16bf a = load_a(Xb, (size_t)s0 * H, H, k0, lane);
    v16bf b = load_b(Wb, (size_t)o0 * H, H, k0, lane);   // B[k][n]=W[o0+n][k]
    acc = wmma_bf16(a, b, acc);
  }
  const int half = lane >> 4, col = lane & 15;
  const int o = o0 + col, h = o >> 6, d = o & 63;
  #pragma unroll
  for (int r = 0; r < 8; ++r) {
    int s = s0 + r + 8 * half;
    float v = acc[r];
    if (z < 2) v = (v > 0.f) ? v + 1.f : __expf(v);       // elu(x)+1
    unsigned short bv = f2bf(v);
    if (z == 0) {
      sqb[((size_t)(h * S + s)) * D + d] = bv;
    } else if (z == 1) {
      skb [((size_t)(h * S + s)) * D + d] = bv;
      skTb[((size_t)(h * D + d)) * S + s] = bv;
    } else {
      vTb [((size_t)(h * D + d)) * S + s] = bv;
    }
  }
}

// ---------------------------------------------------------------- rel mean
__global__ __launch_bounds__(256) void k_relsum(const unsigned short* __restrict__ sqb,
                                                const float* __restrict__ norms,
                                                float* __restrict__ relsum) {
  const int mh = blockIdx.x, m = mh >> 4, h = mh & 15;
  __shared__ float red[256];
  float acc = 0.f;
  const float* nr = norms + (size_t)(m * NH + h) * D;
  for (int s = threadIdx.x; s < S; s += 256) {
    const unsigned short* q = sqb + ((size_t)(h * S + s)) * D;
    float dot = 0.f;
    for (int d = 0; d < D; ++d) dot += bf2f(q[d]) * nr[d];
    acc += dot;
  }
  red[threadIdx.x] = acc;
  __syncthreads();
  for (int st = 128; st > 0; st >>= 1) {
    if (threadIdx.x < st) red[threadIdx.x] += red[threadIdx.x + st];
    __syncthreads();
  }
  if (threadIdx.x == 0) relsum[mh] = red[0];
}

// ---------------------------------------------------------------- memory retrieval
__global__ __launch_bounds__(128) void k_memout(
    const unsigned short* __restrict__ sqb, const unsigned short* __restrict__ memTb,
    const float* __restrict__ norms, const float* __restrict__ relsum,
    float* __restrict__ memout) {
  const int lane = threadIdx.x & 31, wave = threadIdx.x >> 5, tid = threadIdx.x;
  const int s0 = blockIdx.x * 16, h = blockIdx.y;
  __shared__ float relinv[NM][16];
  __shared__ float wm[NM];
  if (tid < 64) {
    int m = tid >> 4, i = tid & 15, s = s0 + i;
    const unsigned short* q = sqb + ((size_t)(h * S + s)) * D;
    const float* nr = norms + (size_t)(m * NH + h) * D;
    float dot = 0.f;
    for (int d = 0; d < D; ++d) dot += bf2f(q[d]) * nr[d];
    relinv[m][i] = 1.f / fmaxf(dot, EPS);
  }
  if (tid < NM) {
    float r0 = relsum[0 * NH + h] / (float)S, r1 = relsum[1 * NH + h] / (float)S;
    float r2 = relsum[2 * NH + h] / (float)S, r3 = relsum[3 * NH + h] / (float)S;
    float mx = fmaxf(fmaxf(r0, r1), fmaxf(r2, r3));
    float e0 = __expf(r0 - mx), e1 = __expf(r1 - mx);
    float e2 = __expf(r2 - mx), e3 = __expf(r3 - mx);
    float inv = 1.f / (e0 + e1 + e2 + e3);
    wm[tid] = (tid == 0 ? e0 : tid == 1 ? e1 : tid == 2 ? e2 : e3) * inv;
  }
  __syncthreads();
  const int et = wave, half = lane >> 4, col = lane & 15;
  v8f out = {};
  for (int m = 0; m < NM; ++m) {
    v8f acc = {};
    for (int k0 = 0; k0 < D; k0 += 32) {
      v16bf a = load_a(sqb, (size_t)(h * S + s0) * D, D, k0, lane);
      v16bf b = load_b(memTb, ((size_t)((m * NH + h) * D) + et * 16) * D, D, k0, lane);
      acc = wmma_bf16(a, b, acc);
    }
    float w = wm[m];
    #pragma unroll
    for (int r = 0; r < 8; ++r) out[r] += w * acc[r] * relinv[m][r + 8 * half];
  }
  #pragma unroll
  for (int r = 0; r < 8; ++r) {
    int s = s0 + r + 8 * half, e = et * 16 + col;
    memout[((size_t)(h * S + s)) * D + e] = out[r];
  }
}

// ---------------------------------------------------------------- chunked causal linear attention
__global__ __launch_bounds__(256) void k_attn(
    const unsigned short* __restrict__ sqb, const unsigned short* __restrict__ skb,
    const unsigned short* __restrict__ skTb, const unsigned short* __restrict__ vTb,
    const float* __restrict__ memout, const float* __restrict__ gate,
    unsigned short* __restrict__ attnb) {
  const int h = blockIdx.x;
  const int tid = threadIdx.x, lane = tid & 31, wave = tid >> 5;
  const int half = lane >> 4, col = lane & 15;
  __shared__ float          Hf[64][65];   // state H[d][e], f32
  __shared__ unsigned short HbT[64][72];  // bf16 H^T [e][d] (B-col layout)
  __shared__ unsigned short Pb[64][72];   // bf16 P  [s][t] (A layout)
  __shared__ float zst[64];
  __shared__ float rowsum[64];
  __shared__ float deninv[64];
  for (int i = tid; i < 64 * 64; i += 256) Hf[i >> 6][i & 63] = 0.f;
  if (tid < 64) zst[tid] = 0.f;
  const float g  = 1.f / (1.f + __expf(-gate[h]));
  const size_t qbase = (size_t)h * S * D;   // [h][s][d] base
  const size_t tbase = (size_t)h * D * S;   // [h][d][s] base
  __syncthreads();
  for (int ch = 0; ch < S / 64; ++ch) {
    const int sb = ch * 64;
    // stage 0: snapshot state to bf16 (transposed), clear row sums
    for (int i = tid; i < 64 * 64; i += 256) {
      int d = i >> 6, e = i & 63;
      HbT[e][d] = f2bf(Hf[d][e]);
    }
    if (tid < 64) rowsum[tid] = 0.f;
    __syncthreads();
    // stage 1: P = tril(SQ @ SK^T), row sums
    for (int t8 = wave; t8 < 16; t8 += 8) {
      int st = t8 >> 2, tt = t8 & 3;
      v8f acc = {};
      for (int k0 = 0; k0 < D; k0 += 32) {
        v16bf a = load_a(sqb, qbase + (size_t)(sb + st * 16) * D, D, k0, lane);
        v16bf b = load_b(skb, qbase + (size_t)(sb + tt * 16) * D, D, k0, lane);
        acc = wmma_bf16(a, b, acc);
      }
      #pragma unroll
      for (int r = 0; r < 8; ++r) {
        int sr = st * 16 + r + 8 * half, tc = tt * 16 + col;
        float v = (tc <= sr) ? acc[r] : 0.f;   // causal mask (same chunk)
        Pb[sr][tc] = f2bf(v);
        atomicAdd(&rowsum[sr], v);
      }
    }
    __syncthreads();
    // stage 2a: den = sq.zstate + rowsum
    if (tid < 64) {
      const unsigned short* q = sqb + qbase + (size_t)(sb + tid) * D;
      float dot = 0.f;
      for (int d = 0; d < D; ++d) dot += bf2f(q[d]) * zst[d];
      deninv[tid] = 1.f / fmaxf(dot + rowsum[tid], EPS);
    }
    __syncthreads();
    // stage 2b: out = (SQ@H + P@V)/den, gate-combine with memout, store bf16
    for (int t8 = wave; t8 < 16; t8 += 8) {
      int st = t8 >> 2, et = t8 & 3;
      v8f acc = {};
      for (int k0 = 0; k0 < D; k0 += 32) {
        v16bf a = load_a(sqb, qbase + (size_t)(sb + st * 16) * D, D, k0, lane);
        v16bf b = load_b(&HbT[0][0], (size_t)et * 16 * 72, 72, k0, lane);
        acc = wmma_bf16(a, b, acc);
      }
      for (int k0 = 0; k0 < 64; k0 += 32) {
        v16bf a = load_a(&Pb[0][0], (size_t)st * 16 * 72, 72, k0, lane);
        v16bf b = load_b(vTb, tbase + (size_t)et * 16 * S + sb, S, k0, lane);
        acc = wmma_bf16(a, b, acc);
      }
      #pragma unroll
      for (int r = 0; r < 8; ++r) {
        int srl = st * 16 + r + 8 * half;
        int s = sb + srl, e = et * 16 + col;
        float local = acc[r] * deninv[srl];
        float mo = memout[qbase + (size_t)s * D + e];
        attnb[(size_t)s * H + h * D + e] = f2bf(g * mo + (1.f - g) * local);
      }
    }
    // stage 3: H += SK^T @ V ; z += colsum(SK)
    for (int t8 = wave; t8 < 16; t8 += 8) {
      int dt = t8 >> 2, et = t8 & 3;
      v8f acc = {};
      for (int k0 = 0; k0 < 64; k0 += 32) {
        v16bf a = load_a(skTb, tbase + (size_t)dt * 16 * S + sb, S, k0, lane);
        v16bf b = load_b(vTb,  tbase + (size_t)et * 16 * S + sb, S, k0, lane);
        acc = wmma_bf16(a, b, acc);
      }
      #pragma unroll
      for (int r = 0; r < 8; ++r)
        Hf[dt * 16 + r + 8 * half][et * 16 + col] += acc[r];
    }
    if (tid < 64) {
      const unsigned short* kt = skTb + tbase + (size_t)tid * S + sb;
      float sum = 0.f;
      for (int t = 0; t < 64; ++t) sum += bf2f(kt[t]);
      zst[tid] += sum;
    }
    __syncthreads();
  }
}

// ---------------------------------------------------------------- output projection
__global__ __launch_bounds__(256) void k_outproj(const unsigned short* __restrict__ Ab,
                                                 const unsigned short* __restrict__ Wob,
                                                 float* __restrict__ out) {
  const int lane = threadIdx.x & 31, wave = threadIdx.x >> 5;
  const int s0 = blockIdx.x * 16;
  const int o0 = blockIdx.y * 128 + wave * 16;
  v8f acc = {};
  for (int k0 = 0; k0 < H; k0 += 32) {
    v16bf a = load_a(Ab, (size_t)s0 * H, H, k0, lane);
    v16bf b = load_b(Wob, (size_t)o0 * H, H, k0, lane);
    acc = wmma_bf16(a, b, acc);
  }
  const int half = lane >> 4, col = lane & 15;
  #pragma unroll
  for (int r = 0; r < 8; ++r) {
    int s = s0 + r + 8 * half;
    out[(size_t)s * H + o0 + col] = acc[r];
  }
}

// ---------------------------------------------------------------- launch
extern "C" void kernel_launch(void* const* d_in, const int* in_sizes, int n_in,
                              void* d_out, int out_size, void* d_ws, size_t ws_size,
                              hipStream_t stream) {
  (void)in_sizes; (void)n_in; (void)out_size; (void)ws_size;
  const float* hs   = (const float*)d_in[0];
  const float* wq   = (const float*)d_in[1];
  const float* wk   = (const float*)d_in[2];
  const float* wv   = (const float*)d_in[3];
  const float* wo   = (const float*)d_in[4];
  const float* gate = (const float*)d_in[5];
  const float* mem  = (const float*)d_in[6];
  const float* mnrm = (const float*)d_in[7];

  char* ws = (char*)d_ws;
  size_t off = 0;
  auto alloc = [&](size_t bytes) {
    void* p = ws + off;
    off += (bytes + 255) & ~(size_t)255;
    return p;
  };
  unsigned short* Xb     = (unsigned short*)alloc((size_t)S * H * 2);
  unsigned short* Wqb    = (unsigned short*)alloc((size_t)H * H * 2);
  unsigned short* Wkb    = (unsigned short*)alloc((size_t)H * H * 2);
  unsigned short* Wvb    = (unsigned short*)alloc((size_t)H * H * 2);
  unsigned short* Wob    = (unsigned short*)alloc((size_t)H * H * 2);
  unsigned short* sqb    = (unsigned short*)alloc((size_t)S * H * 2);
  unsigned short* skb    = (unsigned short*)alloc((size_t)S * H * 2);
  unsigned short* skTb   = (unsigned short*)alloc((size_t)S * H * 2);
  unsigned short* vTb    = (unsigned short*)alloc((size_t)S * H * 2);
  unsigned short* memTb  = (unsigned short*)alloc((size_t)NM * NH * D * D * 2);
  float*          relsum = (float*)alloc((size_t)NM * NH * 4);
  float*          memout = (float*)alloc((size_t)S * H * 4);
  unsigned short* attnb  = (unsigned short*)alloc((size_t)S * H * 2);

  k_f32_to_bf16<<<(S * H + 255) / 256, 256, 0, stream>>>(hs, Xb, S * H);
  k_f32_to_bf16<<<(H * H + 255) / 256, 256, 0, stream>>>(wq, Wqb, H * H);
  k_f32_to_bf16<<<(H * H + 255) / 256, 256, 0, stream>>>(wk, Wkb, H * H);
  k_f32_to_bf16<<<(H * H + 255) / 256, 256, 0, stream>>>(wv, Wvb, H * H);
  k_f32_to_bf16<<<(H * H + 255) / 256, 256, 0, stream>>>(wo, Wob, H * H);
  k_memT<<<(NM * NH * D * D + 255) / 256, 256, 0, stream>>>(mem, memTb);

  k_qkv<<<dim3(S / 16, H / 128, 3), 256, 0, stream>>>(Xb, Wqb, Wkb, Wvb,
                                                      sqb, skb, skTb, vTb);
  k_relsum<<<NM * NH, 256, 0, stream>>>(sqb, mnrm, relsum);
  k_memout<<<dim3(S / 16, NH), 128, 0, stream>>>(sqb, memTb, mnrm, relsum, memout);
  k_attn<<<NH, 256, 0, stream>>>(sqb, skb, skTb, vTb, memout, gate, attnb);
  k_outproj<<<dim3(S / 16, H / 128), 256, 0, stream>>>(attnb, Wob, (float*)d_out);
}
